// MRNNModel_11991548690509
// MI455X (gfx1250) — matrix operand order repaired and checked by reading, not compiled, not run on hardware
//
#include <hip/hip_runtime.h>
#include <hip/hip_bf16.h>

// MRNN: emb gather -> (precomputed gi GEMM) -> sequential GRU scan (WMMA, h in LDS)
// -> condition-select/pack -> decoder GEMM (WMMA, HBM-bound).
// All matmuls use v_wmma_f32_16x16x32_bf16 with f32 accumulation.

#define Bsz   64
#define Tlen  2048
#define Edim  256
#define Hdim  256
#define Vocab 1024
#define HPITCH 264   // 256 + 8 halves padding to spread LDS banks

typedef __bf16 bf16_t;
typedef __attribute__((ext_vector_type(16))) __bf16 v16bf;
typedef __attribute__((ext_vector_type(8)))  float  v8f;

union Frag   { int4 q[2]; v16bf v; };
union Half8  { int4 q;    bf16_t h[8]; };

__device__ __forceinline__ bf16_t f2bf(float x) {
  unsigned u = __builtin_bit_cast(unsigned, x);
  u += 0x7FFFu + ((u >> 16) & 1u);            // round-to-nearest-even
  unsigned short hs = (unsigned short)(u >> 16);
  return __builtin_bit_cast(bf16_t, hs);
}
__device__ __forceinline__ float bf2f(bf16_t b) {
  unsigned u = ((unsigned)__builtin_bit_cast(unsigned short, b)) << 16;
  return __builtin_bit_cast(float, u);
}
__device__ __forceinline__ float sigmoidf_(float x) {
  return 1.0f / (1.0f + __expf(-x));
}
// A-matrix 16-bit 16x32 fragment: lane group g, half j -> local K
__device__ __forceinline__ int a_klocal(int j, int g) { return j + 8 * g + (j & 8); }

__device__ __forceinline__ v8f wmma_bf16(v16bf a, v16bf b, v8f c) {
  return __builtin_amdgcn_wmma_f32_16x16x32_bf16(false, a, false, b, (short)0, c,
                                                 false, false);
}

// ---------------------------------------------------------------------------
// Kernel 1: weights -> bf16, packed in WMMA B-fragment layout (lane-major 32B).
// wcomb: K=512 (k<256 => W_hh cols, k>=256 => W_ih cols), N=768. 48 ntiles x 16 ktiles.
// wdec : K=256, N=1024. 64 ntiles x 8 ktiles.
__global__ void prep_weights_kernel(const float* __restrict__ W_ih,
                                    const float* __restrict__ W_hh,
                                    const float* __restrict__ W_dec,
                                    bf16_t* __restrict__ wcomb,
                                    bf16_t* __restrict__ wdec) {
  int idx = blockIdx.x * blockDim.x + threadIdx.x;
  const int NC = 48 * 16 * 32 * 16;  // 393216 wcomb elements
  if (idx < NC) {
    int j = idx & 15, lane = (idx >> 4) & 31, kt = (idx >> 9) & 15, nt = idx >> 13;
    int n = nt * 16 + (lane & 15);
    int k = kt * 32 + a_klocal(j, lane >> 4);       // B-fragment K mapping
    float v = (k < Hdim) ? W_hh[n * Hdim + k] : W_ih[n * Edim + (k - Hdim)];
    wcomb[idx] = f2bf(v);
  } else {
    int id2 = idx - NC;
    if (id2 < 64 * 8 * 32 * 16) {                   // 262144 wdec elements
      int j = id2 & 15, lane = (id2 >> 4) & 31, kt = (id2 >> 9) & 7, nt = id2 >> 12;
      int n = nt * 16 + (lane & 15);
      int k = kt * 32 + a_klocal(j, lane >> 4);
      wdec[id2] = f2bf(W_dec[n * Hdim + k]);
    }
  }
}

// ---------------------------------------------------------------------------
// Kernel 2: embedding gather, packed in WMMA A-fragment layout.
// xpack[t][mt(4)][kt(8)][lane(32)][j(16)]  (row = b within mt-tile, K = E)
__global__ void embed_pack_kernel(const int* __restrict__ tokens,
                                  const float* __restrict__ emb_table,
                                  bf16_t* __restrict__ xpack) {
  int idx = blockIdx.x * blockDim.x + threadIdx.x;   // T*4*8*32*16 = 33554432
  int j = idx & 15, lane = (idx >> 4) & 31, kt = (idx >> 9) & 7;
  int mt = (idx >> 12) & 3, t = idx >> 14;
  int b = mt * 16 + (lane & 15);
  int e = kt * 32 + a_klocal(j, lane >> 4);
  int tok = tokens[b * Tlen + t];
  xpack[idx] = f2bf(emb_table[tok * Edim + e]);
}

// ---------------------------------------------------------------------------
// Kernel 3: gi = x @ W_ih^T + b_ih for all (b,t) — the time-parallel half of
// the GRU. Output packed per C-tile: gipack[t*4+mt][nt(48)][lane(32)][8] bf16.
__global__ __launch_bounds__(256)
void gi_gemm_kernel(const bf16_t* __restrict__ xpack,
                    const bf16_t* __restrict__ wcomb,
                    const float* __restrict__ b_ih,
                    bf16_t* __restrict__ gipack) {
  int wave = threadIdx.x >> 5, lane = threadIdx.x & 31;
  int tile = blockIdx.x * 8 + wave;                  // 0 .. 2048*4*48-1
  int nt = tile % 48;
  int tm = tile / 48;                                // t*4 + mt
  v8f acc = {};
#pragma unroll
  for (int kt = 0; kt < 8; ++kt) {
    Frag a, b;
    const bf16_t* ap = xpack + (((tm * 8 + kt) * 32 + lane) << 4);
    a.q[0] = ((const int4*)ap)[0];
    a.q[1] = ((const int4*)ap)[1];
    const bf16_t* bp = wcomb + (((nt * 16 + (8 + kt)) * 32 + lane) << 4);
    b.q[0] = ((const int4*)bp)[0];
    b.q[1] = ((const int4*)bp)[1];
    acc = wmma_bf16(a.v, b.v, acc);
  }
  float bias = b_ih[nt * 16 + (lane & 15)];
  Half8 o;
#pragma unroll
  for (int i = 0; i < 8; ++i) o.h[i] = f2bf(acc[i] + bias);
  *(int4*)(gipack + (((size_t)tile * 32 + lane) << 3)) = o.q;
}

// ---------------------------------------------------------------------------
// Kernel 4: sequential GRU scan. One block (32 waves) on one WGP; h lives in
// LDS (bf16, padded). Per step, per wave: 2 tile-triples x (3 gates x 8 K) =
// 48 v_wmma_f32_16x16x32_bf16. gi (input preactivations) streamed from global.
__global__ __launch_bounds__(1024, 1)
void gru_scan_kernel(const bf16_t* __restrict__ gipack,
                     const bf16_t* __restrict__ wcomb,
                     const float* __restrict__ b_hh,
                     bf16_t* __restrict__ hs,
                     float* __restrict__ h_last_out) {
  __shared__ bf16_t Hlds[Bsz][HPITCH];
  int tid = threadIdx.x;
  {
    bf16_t z = __builtin_bit_cast(bf16_t, (unsigned short)0);
    bf16_t* hp = &Hlds[0][0];
    for (int i = tid; i < Bsz * HPITCH; i += 1024) hp[i] = z;
  }
  __syncthreads();

  int wave = tid >> 5, lane = tid & 31;
  int tr0 = wave * 2;
  int mt  = tr0 >> 4;        // same m-tile for both triples of this wave
  int ct0 = tr0 & 15;        // even; second triple uses ct0+1
  int g = lane >> 4, nl = lane & 15;

  float bhh_rzn[2][3];
#pragma unroll
  for (int q = 0; q < 2; ++q) {
    int c = (ct0 + q) * 16 + nl;
    bhh_rzn[q][0] = b_hh[c];
    bhh_rzn[q][1] = b_hh[256 + c];
    bhh_rzn[q][2] = b_hh[512 + c];
  }

  float hnew[2][8];
  for (int t = 0; t < Tlen; ++t) {
#pragma unroll
    for (int q = 0; q < 2; ++q) {
      int ct = ct0 + q;
      // load gi fragments (i_r+b_ih_r, i_z+b_ih_z, i_n+b_ih_n) for this tile
      size_t gbase = ((size_t)(t * 4 + mt) * 48) * 256;  // *32*8
      Half8 gr, gz, gn;
      gr.q = *(const int4*)(gipack + gbase + (((size_t)ct      * 32 + lane) << 3));
      gz.q = *(const int4*)(gipack + gbase + (((size_t)(ct+16) * 32 + lane) << 3));
      gn.q = *(const int4*)(gipack + gbase + (((size_t)(ct+32) * 32 + lane) << 3));

      v8f ar, az, an;
      float gin[8];
#pragma unroll
      for (int i = 0; i < 8; ++i) {
        ar[i] = bf2f(gr.h[i]) + bhh_rzn[q][0];  // i_r + b_ih_r + b_hh_r
        az[i] = bf2f(gz.h[i]) + bhh_rzn[q][1];
        an[i] = bhh_rzn[q][2];                  // h_n accumulates separately
        gin[i] = bf2f(gn.h[i]);                 // i_n + b_ih_n
      }

#pragma unroll
      for (int kt = 0; kt < 8; ++kt) {
        Frag a;
        const bf16_t* rp = &Hlds[mt * 16 + nl][kt * 32 + g * 8];
        a.q[0] = *(const int4*)rp;
        a.q[1] = *(const int4*)(rp + 16);

        Frag br, bz, bn;
        const bf16_t* pr = wcomb + (((ct        * 16 + kt) * 32 + lane) << 4);
        const bf16_t* pz = wcomb + ((((ct + 16) * 16 + kt) * 32 + lane) << 4);
        const bf16_t* pn = wcomb + ((((ct + 32) * 16 + kt) * 32 + lane) << 4);
        br.q[0] = ((const int4*)pr)[0]; br.q[1] = ((const int4*)pr)[1];
        bz.q[0] = ((const int4*)pz)[0]; bz.q[1] = ((const int4*)pz)[1];
        bn.q[0] = ((const int4*)pn)[0]; bn.q[1] = ((const int4*)pn)[1];

        ar = wmma_bf16(a.v, br.v, ar);
        az = wmma_bf16(a.v, bz.v, az);
        an = wmma_bf16(a.v, bn.v, an);
      }

#pragma unroll
      for (int i = 0; i < 8; ++i) {
        float r  = sigmoidf_(ar[i]);
        float z  = sigmoidf_(az[i]);
        float ng = tanhf(gin[i] + r * an[i]);
        int brow = mt * 16 + i + 8 * g;
        float hold = bf2f(Hlds[brow][ct * 16 + nl]);
        hnew[q][i] = (1.0f - z) * ng + z * hold;
      }
    }
    __syncthreads();   // all reads of h_t complete
#pragma unroll
    for (int q = 0; q < 2; ++q) {
      int ct = ct0 + q;
#pragma unroll
      for (int i = 0; i < 8; ++i) {
        int brow = mt * 16 + i + 8 * g;
        bf16_t hb = f2bf(hnew[q][i]);
        Hlds[brow][ct * 16 + nl] = hb;
        hs[(brow * Tlen + t) * Hdim + ct * 16 + nl] = hb;
        if (t == Tlen - 1)
          h_last_out[brow * Hdim + ct * 16 + nl] = hnew[q][i];
      }
    }
    __syncthreads();   // h_{t+1} visible to all waves
  }
}

// ---------------------------------------------------------------------------
// Kernel 5: apply condition-gather (use_prev ? prev_table[tokens[b,cond]] : hs)
// and pack the decoder A-matrix in WMMA A-fragment layout.
__global__ void select_pack_kernel(const int* __restrict__ tokens,
                                   const int* __restrict__ conditions,
                                   const float* __restrict__ prev_table,
                                   const bf16_t* __restrict__ hs,
                                   bf16_t* __restrict__ apack) {
  int idx = blockIdx.x * blockDim.x + threadIdx.x;   // 8192*8*32*16
  int j = idx & 15, lane = (idx >> 4) & 31, kt = (idx >> 9) & 7, rt = idx >> 12;
  int row = rt * 16 + (lane & 15);                   // row = b*T + t
  int b = row >> 11, t = row & (Tlen - 1);
  int k = kt * 32 + a_klocal(j, lane >> 4);
  int cn = (t + 1 < Tlen) ? conditions[b * Tlen + t + 1] : -1;
  bool use_prev = (t > 4) && (cn != -1);
  float v;
  if (use_prev) {
    int ci = cn < 0 ? 0 : (cn > Tlen - 1 ? Tlen - 1 : cn);
    int tok = tokens[b * Tlen + ci];
    v = prev_table[tok * Hdim + k];
  } else {
    v = bf2f(hs[row * Hdim + k]);
  }
  apack[idx] = f2bf(v);
}

// ---------------------------------------------------------------------------
// Kernel 6: decoder GEMM (HBM-bound, 512 MB f32 out). One 16x16 tile per wave,
// K=256 in 8 bf16 WMMAs; weights (512 KB) stay L2-resident.
__global__ __launch_bounds__(256)
void decoder_kernel(const bf16_t* __restrict__ apack,
                    const bf16_t* __restrict__ wdec,
                    const float* __restrict__ b_dec,
                    float* __restrict__ dec) {
  int wave = threadIdx.x >> 5, lane = threadIdx.x & 31;
  int tileId = blockIdx.x * 8 + wave;   // 8192 mtiles x 64 ntiles
  int mtile = tileId >> 6;
  int nt = tileId & 63;
  if (mtile + 1 < 8192)
    __builtin_prefetch(apack + ((((mtile + 1) * 8) * 32 + lane) << 4), 0, 0);
  v8f acc = {};
#pragma unroll
  for (int kt = 0; kt < 8; ++kt) {
    Frag a, b;
    const bf16_t* ap = apack + (((mtile * 8 + kt) * 32 + lane) << 4);
    a.q[0] = ((const int4*)ap)[0];
    a.q[1] = ((const int4*)ap)[1];
    const bf16_t* bp = wdec + (((nt * 8 + kt) * 32 + lane) << 4);
    b.q[0] = ((const int4*)bp)[0];
    b.q[1] = ((const int4*)bp)[1];
    acc = wmma_bf16(a.v, b.v, acc);
  }
  int nl = lane & 15, g = lane >> 4;
  int n = nt * 16 + nl;
  float bias = b_dec[n];
#pragma unroll
  for (int i = 0; i < 8; ++i) {
    int row = mtile * 16 + i + 8 * g;
    dec[row * Vocab + n] = acc[i] + bias;
  }
}

// ---------------------------------------------------------------------------
extern "C" void kernel_launch(void* const* d_in, const int* in_sizes, int n_in,
                              void* d_out, int out_size, void* d_ws, size_t ws_size,
                              hipStream_t stream) {
  (void)in_sizes; (void)n_in; (void)out_size; (void)ws_size;
  const int*   tokens     = (const int*)d_in[0];
  const int*   conditions = (const int*)d_in[1];
  const float* emb_table  = (const float*)d_in[2];
  const float* prev_table = (const float*)d_in[3];
  const float* W_ih       = (const float*)d_in[4];
  const float* b_ih       = (const float*)d_in[5];
  const float* W_hh       = (const float*)d_in[6];
  const float* b_hh       = (const float*)d_in[7];
  const float* W_dec      = (const float*)d_in[8];
  const float* b_dec      = (const float*)d_in[9];

  float* dec    = (float*)d_out;
  float* h_last = dec + (size_t)Bsz * Tlen * Vocab;

  char* p = (char*)d_ws;
  bf16_t* wcomb  = (bf16_t*)p; p += (size_t)512 * 768 * 2;        // 768 KB
  bf16_t* wdec   = (bf16_t*)p; p += (size_t)256 * 1024 * 2;       // 512 KB
  bf16_t* xbuf   = (bf16_t*)p; p += (size_t)33554432 * 2;         // 64 MB (xpack, later apack)
  bf16_t* gipack = (bf16_t*)p; p += (size_t)100663296 * 2;        // 192 MB
  bf16_t* hs     = (bf16_t*)p; p += (size_t)33554432 * 2;         // 64 MB
  bf16_t* apack  = xbuf;  // xpack dead after gi_gemm; reuse region

  prep_weights_kernel<<<(655360 + 255) / 256, 256, 0, stream>>>(W_ih, W_hh, W_dec,
                                                                wcomb, wdec);
  embed_pack_kernel<<<33554432 / 256, 256, 0, stream>>>(tokens, emb_table, xbuf);
  gi_gemm_kernel<<<49152, 256, 0, stream>>>(xbuf, wcomb, b_ih, gipack);
  gru_scan_kernel<<<1, 1024, 0, stream>>>(gipack, wcomb, b_hh, hs, h_last);
  select_pack_kernel<<<33554432 / 256, 256, 0, stream>>>(tokens, conditions,
                                                         prev_table, hs, apack);
  decoder_kernel<<<65536, 256, 0, stream>>>(apack, wdec, b_dec, dec);
}
